// Coref_41334765257269
// MI455X (gfx1250) — matrix operand order repaired and checked by reading, not compiled, not run on hardware
//
#include <hip/hip_runtime.h>
#include <math.h>

#define WTOK 2048
#define H    768
#define MSW  10
#define WFD  20
#define DD   2324          // 3*H + WF
#define DP   2336          // DD padded to 32
#define D2   4648          // 2*DD
#define D2P  4672          // padded
#define MSEL 819           // int(0.4*2048)
#define MP   832           // MSEL padded to 16
#define KANT 50
#define NCAND (WTOK*MSW)   // 20480
#define NEGF (-1e30f)

#define OFF_TS   0
#define OFF_TE   (MSEL)
#define OFF_TA   (2*MSEL)
#define OFF_FS   (2*MSEL + MSEL*KANT)
#define OFF_LOSS (2*MSEL + MSEL*KANT + MSEL*(KANT+1))

typedef _Float16 f16;
typedef __attribute__((ext_vector_type(16))) _Float16 v16h;
typedef __attribute__((ext_vector_type(8)))  _Float16 v8h;
typedef __attribute__((ext_vector_type(8)))  float    v8f;
union V16H { v16h v; v8h q[2]; _Float16 h[16]; };

__device__ __forceinline__ float wave_sum(float x) {
    for (int off = 16; off; off >>= 1) x += __shfl_xor(x, off, 32);
    return x;
}

__device__ __forceinline__ unsigned int fkey(float f) {
    unsigned int u = __float_as_uint(f);
    return (u & 0x80000000u) ? ~u : (u | 0x80000000u);
}

// ---------------- WMMA fragment loaders (16x16x32 f16, wave32 layouts) -----
// A (16x32, row-major): lane holds row (lane&15); elements live at
// k0 + 8*hi + [0..7] and k0 + 16 + 8*hi + [0..7]  -> two b128 loads.
__device__ __forceinline__ v16h load_a_frag(const f16* A, int lda, int row, int k0, int hi) {
    const f16* Ar = A + (size_t)row * lda + k0 + 8 * hi;
    V16H a;
    a.q[0] = *(const v8h*)(Ar);
    a.q[1] = *(const v8h*)(Ar + 16);
    return a.v;
}
// B (32x16) from a K-contiguous (transposed) buffer: Bt[n][k] = B[k][n].
// lane holds col (lane&15); elements are k0 + 8*hi + [0..7] and k0 + 16 + 8*hi + [0..7] -> two b128s.
__device__ __forceinline__ v16h load_bt_frag(const f16* Bt, int ldbt, int col, int k0, int hi) {
    const f16* Br = Bt + (size_t)col * ldbt + k0 + 8 * hi;
    V16H b;
    b.q[0] = *(const v8h*)(Br);
    b.q[1] = *(const v8h*)(Br + 16);
    return b.v;
}

// ---------------- K0: zero X (pad rows must stay zero) ---------------------
__global__ void zero_x_kernel(f16* Xb) {
    int i = blockIdx.x * blockDim.x + threadIdx.x;
    if (i < MP * D2P) Xb[i] = (f16)0.0f;
}

// ---------------- K1: per-token stats (alphas, s1..s3) ---------------------
__global__ void token_stats_kernel(const float* __restrict__ seq,
                                   const float* __restrict__ W_alpha,
                                   const float* __restrict__ b_alpha,
                                   const float* __restrict__ W_m,
                                   float* alphas, float* s1, float* s2, float* s3) {
    int gid = blockIdx.x * blockDim.x + threadIdx.x;
    int tok = gid >> 5, lane = gid & 31;
    if (tok >= WTOK) return;
    float a0 = 0.f, a1 = 0.f, a2 = 0.f, a3 = 0.f;
    const float* row = seq + (size_t)tok * H;
    for (int d = lane; d < H; d += 32) {
        float s = row[d];
        a0 += s * W_alpha[d];
        a1 += s * W_m[d];
        a2 += s * W_m[H + d];
        a3 += s * W_m[2 * H + d];
    }
    a0 = wave_sum(a0); a1 = wave_sum(a1); a2 = wave_sum(a2); a3 = wave_sum(a3);
    if (lane == 0) {
        alphas[tok] = a0 + b_alpha[0];
        s1[tok] = a1; s2[tok] = a2; s3[tok] = a3;
    }
}

// ---------------- K2: mention scores per candidate -------------------------
__global__ void mention_kernel(const float* __restrict__ alphas,
                               const float* __restrict__ s1,
                               const float* __restrict__ s2,
                               const float* __restrict__ s3,
                               const float* __restrict__ W_m,
                               const float* __restrict__ b_m,
                               const float* __restrict__ wfeat,
                               const int* __restrict__ smap,
                               const int* __restrict__ gstart,
                               const int* __restrict__ gend,
                               const int* __restrict__ gcid, int G,
                               float* mscore, float* mmask, int* ccl) {
    int c = blockIdx.x * blockDim.x + threadIdx.x;
    if (c >= NCAND) return;
    int cs = c / MSW, off = c % MSW;
    int ce = cs + off;
    int cec = ce < (WTOK - 1) ? ce : (WTOK - 1);
    bool valid = (ce < WTOK) && (smap[cs] == smap[cec]);
    int L = cec - cs + 1;
    float mx = -3e38f;
    for (int j = 0; j < L; ++j) mx = fmaxf(mx, alphas[cs + j]);
    float sum = 0.f, sacc = 0.f;
    for (int j = 0; j < L; ++j) {
        float w = expf(alphas[cs + j] - mx);
        sum += w; sacc += w * s3[cs + j];
    }
    float span = sacc / sum;
    int wdt = (cec - cs) < (MSW - 1) ? (cec - cs) : (MSW - 1);
    float wsc = 0.f;
    for (int f = 0; f < WFD; ++f) wsc += wfeat[wdt * WFD + f] * W_m[3 * H + f];
    float sc = s1[cs] + s2[cec] + span + wsc + b_m[0];
    mscore[c] = sc;
    mmask[c] = valid ? sc : NEGF;
    int cc = 0;
    for (int g = 0; g < G; ++g)
        if (cs == gstart[g] && cec == gend[g]) cc += gcid[g];
    ccl[c] = cc;
}

// ---------------- K3: radix-select top-819 + index-ordered compaction ------
__global__ void topk_mention_kernel(const float* __restrict__ mmask,
                                    const float* __restrict__ mscore,
                                    const int* __restrict__ ccl,
                                    int* tsA, int* teA, float* tscoreA,
                                    int* tclA, int* twA, float* dout) {
    __shared__ unsigned int red[1024];
    __shared__ unsigned int sA[1024];
    __shared__ unsigned int sE[1024];
    const int tid = threadIdx.x;
    const int CH = NCAND / 1024;  // 20
    unsigned int key[CH];
#pragma unroll
    for (int e = 0; e < CH; ++e) key[e] = fkey(mmask[tid * CH + e]);

    // threshold T = 819th-largest key (bitwise build on monotone uint key)
    unsigned int T = 0;
    for (int b = 31; b >= 0; --b) {
        unsigned int cand = T | (1u << b);
        unsigned int c = 0;
#pragma unroll
        for (int e = 0; e < CH; ++e) c += (key[e] >= cand) ? 1u : 0u;
        red[tid] = c; __syncthreads();
        for (int s = 512; s > 0; s >>= 1) { if (tid < s) red[tid] += red[tid + s]; __syncthreads(); }
        unsigned int tot = red[0]; __syncthreads();
        if (tot >= MSEL) T = cand;
    }
    {
        unsigned int c = 0;
#pragma unroll
        for (int e = 0; e < CH; ++e) c += (key[e] > T) ? 1u : 0u;
        red[tid] = c; __syncthreads();
        for (int s = 512; s > 0; s >>= 1) { if (tid < s) red[tid] += red[tid + s]; __syncthreads(); }
    }
    unsigned int cGT = red[0]; __syncthreads();
    unsigned int extra = MSEL - cGT;   // ties at T: lowest indices win

    unsigned int aCnt = 0, eCnt = 0;
#pragma unroll
    for (int e = 0; e < CH; ++e) { aCnt += (key[e] > T); eCnt += (key[e] == T); }
    sA[tid] = aCnt; sE[tid] = eCnt; __syncthreads();
    if (tid == 0) {
        unsigned int ra = 0, re = 0;
        for (int t = 0; t < 1024; ++t) {
            unsigned int a = sA[t], e = sE[t];
            sA[t] = ra; sE[t] = re; ra += a; re += e;
        }
    }
    __syncthreads();

    unsigned int a = sA[tid], e = sE[tid];
    for (int el = 0; el < CH; ++el) {
        int i = tid * CH + el;
        unsigned int ky = key[el];
        unsigned int pos = 0xFFFFFFFFu;
        if (ky > T) { pos = a + (e < extra ? e : extra); a++; }
        else if (ky == T) { if (e < extra) pos = a + e; e++; }
        if (pos < MSEL) {
            int cs = i / MSW;
            int ce = cs + (i % MSW);
            int cec = ce < (WTOK - 1) ? ce : (WTOK - 1);
            tsA[pos] = cs; teA[pos] = cec;
            int wdt = (cec - cs) < (MSW - 1) ? (cec - cs) : (MSW - 1);
            twA[pos] = wdt;
            tscoreA[pos] = mscore[i];
            tclA[pos] = ccl[i];
            dout[OFF_TS + pos] = (float)cs;
            dout[OFF_TE + pos] = (float)cec;
        }
    }
}

// ---------------- K4: build padded reps (f32 + f16) ------------------------
__global__ void build_rep_kernel(const float* __restrict__ seq,
                                 const float* __restrict__ alphas,
                                 const float* __restrict__ wfeat,
                                 const int* __restrict__ tsA,
                                 const int* __restrict__ teA,
                                 const int* __restrict__ twA,
                                 float* repsF, f16* repsH) {
    int r = blockIdx.x;
    if (r >= MSEL) {
        for (int col = threadIdx.x; col < DP; col += blockDim.x) {
            repsF[(size_t)r * DP + col] = 0.f;
            repsH[(size_t)r * DP + col] = (f16)0.0f;
        }
        return;
    }
    __shared__ float wls[MSW];
    __shared__ int shTs, shTe, shL, shW;
    if (threadIdx.x == 0) {
        int ts = tsA[r], te = teA[r];
        shTs = ts; shTe = te; shW = twA[r];
        int L = te - ts + 1; shL = L;
        float mx = -3e38f;
        for (int j = 0; j < L; ++j) mx = fmaxf(mx, alphas[ts + j]);
        float s = 0.f;
        for (int j = 0; j < L; ++j) { float w = expf(alphas[ts + j] - mx); wls[j] = w; s += w; }
        float inv = 1.f / s;
        for (int j = 0; j < L; ++j) wls[j] *= inv;
    }
    __syncthreads();
    int ts = shTs, te = shTe, L = shL, wdt = shW;
    for (int col = threadIdx.x; col < DP; col += blockDim.x) {
        float v;
        if (col < H)           v = seq[(size_t)ts * H + col];
        else if (col < 2 * H)  v = seq[(size_t)te * H + (col - H)];
        else if (col < 3 * H) {
            int d = col - 2 * H;
            float acc = 0.f;
            for (int j = 0; j < L; ++j) acc += wls[j] * seq[(size_t)(ts + j) * H + d];
            v = acc;
        } else if (col < DD)   v = wfeat[wdt * WFD + (col - 3 * H)];
        else                   v = 0.f;
        repsF[(size_t)r * DP + col] = v;
        repsH[(size_t)r * DP + col] = (f16)v;
    }
}

// ---------------- K5: weight conversions to TRANSPOSED padded f16 ----------
// WcT[n][k] = W_c[k][n]   (DP x DP, ld = DP)
__global__ void convert_wc_kernel(const float* __restrict__ W_c, f16* WcT) {
    int idx = blockIdx.x * blockDim.x + threadIdx.x;
    if (idx >= DP * DP) return;
    int n = idx / DP, k = idx % DP;
    float v = (k < DD && n < DD) ? W_c[(size_t)k * DD + n] : 0.f;
    WcT[idx] = (f16)v;
}
// WfT[n][k] = Wf_padded[k][n]  (DP x D2P, ld = D2P); padded k-rows:
// k<2324 -> W_f[k], 2324..2335 -> 0, 2336..4659 -> W_f[k-12], else 0
__global__ void convert_wf_kernel(const float* __restrict__ W_f, f16* WfT) {
    int idx = blockIdx.x * blockDim.x + threadIdx.x;
    if (idx >= DP * D2P) return;
    int n = idx / D2P, k = idx % D2P;
    float v = 0.f;
    if (n < DD) {
        if (k < DD) v = W_f[(size_t)k * DD + n];
        else if (k >= DP) {
            int kp = k - (DP - DD);
            if (kp < D2) v = W_f[(size_t)kp * DD + n];
        }
    }
    WfT[idx] = (f16)v;
}

// ---------------- K6: U = reps @ W_c  (WMMA, 2-wide N) ---------------------
__global__ void gemm_rep_wc_kernel(const f16* __restrict__ A,
                                   const f16* __restrict__ Bt, f16* U) {
    int wave = (blockIdx.x * blockDim.x + threadIdx.x) >> 5;
    int lane = threadIdx.x & 31;
    const int tilesN2 = (DP / 16) / 2;   // 73
    const int tilesM  = MP / 16;         // 52
    if (wave >= tilesM * tilesN2) return;
    int tm = wave / tilesN2, tn = (wave % tilesN2) * 2;
    int r16 = lane & 15, hi = (lane >> 4) & 1;
    int row = tm * 16 + r16, c0 = tn * 16 + r16, c1 = c0 + 16;
    v8f acc0 = {}, acc1 = {};
    for (int kt = 0; kt < DP / 32; ++kt) {
        int k0 = kt * 32;
        v16h a  = load_a_frag(A, DP, row, k0, hi);
        v16h b0 = load_bt_frag(Bt, DP, c0, k0, hi);
        v16h b1 = load_bt_frag(Bt, DP, c1, k0, hi);
        if (kt + 1 < DP / 32) {
            __builtin_prefetch((const void*)(Bt + (size_t)c0 * DP + k0 + 32), 0, 1);
            __builtin_prefetch((const void*)(Bt + (size_t)c1 * DP + k0 + 32), 0, 1);
        }
        acc0 = __builtin_amdgcn_wmma_f32_16x16x32_f16(false, a, false, b0, (short)0, acc0, false, false);
        acc1 = __builtin_amdgcn_wmma_f32_16x16x32_f16(false, a, false, b1, (short)0, acc1, false, false);
    }
#pragma unroll
    for (int vv = 0; vv < 8; ++vv) {
        int orow = tm * 16 + vv + 8 * hi;
        U[(size_t)orow * DP + c0] = (f16)acc0[vv];
        U[(size_t)orow * DP + c1] = (f16)acc1[vv];
    }
}

// ---------------- K7: fast = U @ repsT + scores + mask  (WMMA, 2-wide) -----
__global__ void gemm_fast_kernel(const f16* __restrict__ U,
                                 const f16* __restrict__ R,
                                 const float* __restrict__ tscore,
                                 float* fastM) {
    int wave = (blockIdx.x * blockDim.x + threadIdx.x) >> 5;
    int lane = threadIdx.x & 31;
    const int tilesN2 = (MP / 16) / 2;   // 26
    const int tilesM  = MP / 16;         // 52
    if (wave >= tilesM * tilesN2) return;
    int tm = wave / tilesN2, tn = (wave % tilesN2) * 2;
    int r16 = lane & 15, hi = (lane >> 4) & 1;
    int row = tm * 16 + r16, c0 = tn * 16 + r16, c1 = c0 + 16;
    v8f acc0 = {}, acc1 = {};
    for (int kt = 0; kt < DP / 32; ++kt) {
        int k0 = kt * 32;
        v16h a  = load_a_frag(U, DP, row, k0, hi);
        v16h b0 = load_bt_frag(R, DP, c0, k0, hi);
        v16h b1 = load_bt_frag(R, DP, c1, k0, hi);
        acc0 = __builtin_amdgcn_wmma_f32_16x16x32_f16(false, a, false, b0, (short)0, acc0, false, false);
        acc1 = __builtin_amdgcn_wmma_f32_16x16x32_f16(false, a, false, b1, (short)0, acc1, false, false);
    }
#pragma unroll
    for (int vv = 0; vv < 8; ++vv) {
        int orow = tm * 16 + vv + 8 * hi;
        float v0 = NEGF, v1 = NEGF;
        if (orow < MSEL && c0 < MSEL)
            v0 = acc0[vv] + tscore[orow] + tscore[c0] + ((c0 < orow) ? 0.f : NEGF);
        if (orow < MSEL && c1 < MSEL)
            v1 = acc1[vv] + tscore[orow] + tscore[c1] + ((c1 < orow) ? 0.f : NEGF);
        fastM[(size_t)orow * MP + c0] = v0;
        fastM[(size_t)orow * MP + c1] = v1;
    }
}

// ---------------- K8: per-row top-50 antecedents ---------------------------
__global__ void topk_ante_kernel(const float* __restrict__ fastM,
                                 int* top_ante, float* ta_fast, float* dout) {
    int i = blockIdx.x;               // < MSEL
    int lane = threadIdx.x;           // 32 threads
    __shared__ float vals[MSEL];
    for (int j = lane; j < MSEL; j += 32) vals[j] = fastM[(size_t)i * MP + j];
    __syncthreads();
    for (int t = 0; t < KANT; ++t) {
        float bv = -3.0e38f; int bi = 0x3FFFFFFF;
        for (int j = lane; j < MSEL; j += 32) {
            float v = vals[j];
            if (v > bv || (v == bv && j < bi)) { bv = v; bi = j; }
        }
        for (int off = 16; off; off >>= 1) {
            float ov = __shfl_xor(bv, off, 32);
            int   oi = __shfl_xor(bi, off, 32);
            if (ov > bv || (ov == bv && oi < bi)) { bv = ov; bi = oi; }
        }
        if (lane == 0) {
            top_ante[i * KANT + t] = bi;
            ta_fast[i * KANT + t]  = bv;
            dout[OFF_TA + i * KANT + t] = (float)bi;
            vals[bi] = -3.0e38f;
        }
        __syncthreads();
    }
}

// ---------------- K9: slow pair scores (wave per pair) ---------------------
__global__ void slow_kernel(const float* __restrict__ repsF,
                            const float* __restrict__ W_a,
                            const float* __restrict__ b_a,
                            const int* __restrict__ top_ante,
                            const float* __restrict__ ta_fast,
                            float* scores_a) {
    int gw = (blockIdx.x * blockDim.x + threadIdx.x) >> 5;
    if (gw >= MSEL * KANT) return;
    int lane = threadIdx.x & 31;
    int i = gw / KANT, a = gw % KANT;
    int j = top_ante[i * KANT + a];
    const float* ti = repsF + (size_t)i * DP;
    const float* tj = repsF + (size_t)j * DP;
    float acc = 0.f;
    for (int d = lane; d < DD; d += 32) {
        float t = ti[d], an = tj[d];
        acc += t * W_a[d] + an * W_a[DD + d] + an * t * W_a[2 * DD + d];
    }
    acc = wave_sum(acc);
    if (lane == 0) scores_a[i * KANT + a] = ta_fast[i * KANT + a] + acc + b_a[0];
}

// ---------------- K10: softmax + attended + build X ------------------------
__global__ void attend_kernel(const float* __restrict__ repsF,
                              const float* __restrict__ scores_a,
                              const int* __restrict__ top_ante,
                              float* attended, f16* Xb) {
    int i = blockIdx.x;  // < MSEL
    int t = threadIdx.x;
    __shared__ float wts[KANT + 1];
    __shared__ int ja[KANT];
    if (t < KANT) ja[t] = top_ante[i * KANT + t];
    if (t == 0) {
        float mx = 0.f;
        for (int a = 0; a < KANT; ++a) mx = fmaxf(mx, scores_a[i * KANT + a]);
        float s = expf(0.f - mx); wts[0] = s; float sum = s;
        for (int a = 0; a < KANT; ++a) {
            float w = expf(scores_a[i * KANT + a] - mx);
            wts[a + 1] = w; sum += w;
        }
        float inv = 1.f / sum;
        for (int a = 0; a <= KANT; ++a) wts[a] *= inv;
    }
    __syncthreads();
    for (int col = t; col < DP; col += blockDim.x) {
        float rv = repsF[(size_t)i * DP + col];
        float acc = wts[0] * rv;
        for (int a = 0; a < KANT; ++a)
            acc += wts[a + 1] * repsF[(size_t)ja[a] * DP + col];
        attended[(size_t)i * DP + col] = acc;
        Xb[(size_t)i * D2P + col]      = (f16)rv;
        Xb[(size_t)i * D2P + DP + col] = (f16)acc;
    }
}

// ---------------- K11: gate GEMM + fused sigmoid update (WMMA, 2-wide) -----
__global__ void gemm_gate_kernel(const f16* __restrict__ X,
                                 const f16* __restrict__ WfT,
                                 const float* __restrict__ b_f,
                                 const float* __restrict__ attended,
                                 float* repsF, f16* repsH) {
    int wave = (blockIdx.x * blockDim.x + threadIdx.x) >> 5;
    int lane = threadIdx.x & 31;
    const int tilesN2 = (DP / 16) / 2;   // 73
    const int tilesM  = MP / 16;         // 52
    if (wave >= tilesM * tilesN2) return;
    int tm = wave / tilesN2, tn = (wave % tilesN2) * 2;
    int r16 = lane & 15, hi = (lane >> 4) & 1;
    int row = tm * 16 + r16, c0 = tn * 16 + r16, c1 = c0 + 16;
    v8f acc0 = {}, acc1 = {};
    for (int kt = 0; kt < D2P / 32; ++kt) {
        int k0 = kt * 32;
        v16h a  = load_a_frag(X, D2P, row, k0, hi);
        v16h b0 = load_bt_frag(WfT, D2P, c0, k0, hi);
        v16h b1 = load_bt_frag(WfT, D2P, c1, k0, hi);
        if (kt + 1 < D2P / 32) {
            __builtin_prefetch((const void*)(WfT + (size_t)c0 * D2P + k0 + 32), 0, 1);
            __builtin_prefetch((const void*)(WfT + (size_t)c1 * D2P + k0 + 32), 0, 1);
        }
        acc0 = __builtin_amdgcn_wmma_f32_16x16x32_f16(false, a, false, b0, (short)0, acc0, false, false);
        acc1 = __builtin_amdgcn_wmma_f32_16x16x32_f16(false, a, false, b1, (short)0, acc1, false, false);
    }
#pragma unroll
    for (int vv = 0; vv < 8; ++vv) {
        int orow = tm * 16 + vv + 8 * hi;
        float o0 = 0.f, o1 = 0.f;
        if (orow < MSEL && c0 < DD) {
            float g  = 1.f / (1.f + expf(-(acc0[vv] + b_f[c0])));
            float av = attended[(size_t)orow * DP + c0];
            float ov = repsF[(size_t)orow * DP + c0];
            o0 = g * av + (1.f - g) * ov;
        }
        if (orow < MSEL && c1 < DD) {
            float g  = 1.f / (1.f + expf(-(acc1[vv] + b_f[c1])));
            float av = attended[(size_t)orow * DP + c1];
            float ov = repsF[(size_t)orow * DP + c1];
            o1 = g * av + (1.f - g) * ov;
        }
        repsF[(size_t)orow * DP + c0] = o0;
        repsH[(size_t)orow * DP + c0] = (f16)o0;
        repsF[(size_t)orow * DP + c1] = o1;
        repsH[(size_t)orow * DP + c1] = (f16)o1;
    }
}

// ---------------- K12: final scores, labels, per-row loss ------------------
__global__ void final_kernel(const float* __restrict__ scores_a,
                             const int* __restrict__ top_ante,
                             const int* __restrict__ tclA,
                             float* lossRows, float* dout) {
    int i = blockIdx.x;   // < MSEL
    int t = threadIdx.x;  // 64 threads
    __shared__ float fs[KANT + 1];
    __shared__ int lab[KANT + 1];
    if (t == 0) fs[0] = 0.f;
    if (t < KANT) {
        int j = top_ante[i * KANT + t];
        fs[t + 1] = scores_a[i * KANT + t];
        int tc = tclA[i];
        int ac = (j < i) ? tclA[j] : -1;
        lab[t + 1] = (ac == tc && tc > 0) ? 1 : 0;
    }
    __syncthreads();
    if (t == 0) {
        int any = 0;
        for (int a = 1; a <= KANT; ++a) any |= lab[a];
        lab[0] = any ? 0 : 1;
        float mx = fs[0];
        for (int a = 1; a <= KANT; ++a) mx = fmaxf(mx, fs[a]);
        float s = 0.f;
        for (int a = 0; a <= KANT; ++a) s += expf(fs[a] - mx);
        float l1 = mx + logf(s);
        float mg = -3e38f;
        for (int a = 0; a <= KANT; ++a) if (lab[a]) mg = fmaxf(mg, fs[a]);
        float sg = 0.f;
        for (int a = 0; a <= KANT; ++a) if (lab[a]) sg += expf(fs[a] - mg);
        float l2 = mg + logf(sg);
        lossRows[i] = l1 - l2;
    }
    if (t <= KANT) dout[OFF_FS + i * (KANT + 1) + t] = fs[t];
}

__global__ void loss_sum_kernel(const float* __restrict__ lossRows, float* dout) {
    if (blockIdx.x == 0 && threadIdx.x == 0) {
        float s = 0.f;
        for (int i = 0; i < MSEL; ++i) s += lossRows[i];
        dout[OFF_LOSS] = s;
    }
}

// ===========================================================================
extern "C" void kernel_launch(void* const* d_in, const int* in_sizes, int n_in,
                              void* d_out, int out_size, void* d_ws, size_t ws_size,
                              hipStream_t stream) {
    const float* seq     = (const float*)d_in[0];
    const float* W_alpha = (const float*)d_in[1];
    const float* b_alpha = (const float*)d_in[2];
    const float* W_m     = (const float*)d_in[3];
    const float* b_m     = (const float*)d_in[4];
    const float* W_c     = (const float*)d_in[5];
    const float* W_a     = (const float*)d_in[6];
    const float* b_a     = (const float*)d_in[7];
    const float* W_f     = (const float*)d_in[8];
    const float* b_f     = (const float*)d_in[9];
    const float* wfeat   = (const float*)d_in[10];
    const int*   smap    = (const int*)d_in[11];
    const int*   gstart  = (const int*)d_in[12];
    const int*   gend    = (const int*)d_in[13];
    const int*   gcid    = (const int*)d_in[14];
    const int G = in_sizes[12];
    float* dout = (float*)d_out;

    uintptr_t p = (uintptr_t)d_ws;
    auto alloc = [&](size_t bytes) -> void* {
        void* r = (void*)p;
        p += (bytes + 255) & ~(size_t)255;
        return r;
    };
    float* alphas  = (float*)alloc(WTOK * 4);
    float* s1      = (float*)alloc(WTOK * 4);
    float* s2      = (float*)alloc(WTOK * 4);
    float* s3      = (float*)alloc(WTOK * 4);
    float* mscore  = (float*)alloc(NCAND * 4);
    float* mmask   = (float*)alloc(NCAND * 4);
    int*   ccl     = (int*)alloc(NCAND * 4);
    int*   tsA     = (int*)alloc(MP * 4);
    int*   teA     = (int*)alloc(MP * 4);
    float* tscoreA = (float*)alloc(MP * 4);
    int*   tclA    = (int*)alloc(MP * 4);
    int*   twA     = (int*)alloc(MP * 4);
    float* repsF   = (float*)alloc((size_t)MP * DP * 4);
    f16*   repsH   = (f16*)alloc((size_t)MP * DP * 2);
    f16*   WcT     = (f16*)alloc((size_t)DP * DP * 2);
    f16*   WfT     = (f16*)alloc((size_t)DP * D2P * 2);
    f16*   Ub      = (f16*)alloc((size_t)MP * DP * 2);
    float* fastM   = (float*)alloc((size_t)MP * MP * 4);
    int*   top_ante= (int*)alloc(MSEL * KANT * 4);
    float* ta_fast = (float*)alloc(MSEL * KANT * 4);
    float* scores_a= (float*)alloc(MSEL * KANT * 4);
    float* attended= (float*)alloc((size_t)MP * DP * 4);
    f16*   Xb      = (f16*)alloc((size_t)MP * D2P * 2);
    float* lossRows= (float*)alloc(MP * 4);

    // --- setup ---
    zero_x_kernel<<<(MP * D2P + 255) / 256, 256, 0, stream>>>(Xb);
    token_stats_kernel<<<(WTOK * 32) / 256, 256, 0, stream>>>(
        seq, W_alpha, b_alpha, W_m, alphas, s1, s2, s3);
    mention_kernel<<<NCAND / 256, 256, 0, stream>>>(
        alphas, s1, s2, s3, W_m, b_m, wfeat, smap, gstart, gend, gcid, G,
        mscore, mmask, ccl);
    topk_mention_kernel<<<1, 1024, 0, stream>>>(
        mmask, mscore, ccl, tsA, teA, tscoreA, tclA, twA, dout);
    build_rep_kernel<<<MP, 256, 0, stream>>>(
        seq, alphas, wfeat, tsA, teA, twA, repsF, repsH);
    convert_wc_kernel<<<(DP * DP + 255) / 256, 256, 0, stream>>>(W_c, WcT);
    convert_wf_kernel<<<(DP * D2P + 255) / 256, 256, 0, stream>>>(W_f, WfT);

    // --- bilinear antecedent scores (WMMA) ---
    const int wavesBig = (MP / 16) * ((DP / 16) / 2);  // 52*73
    const int wavesSq  = (MP / 16) * ((MP / 16) / 2);  // 52*26
    gemm_rep_wc_kernel<<<(wavesBig + 7) / 8, 256, 0, stream>>>(repsH, WcT, Ub);
    gemm_fast_kernel<<<(wavesSq + 7) / 8, 256, 0, stream>>>(Ub, repsH, tscoreA, fastM);
    topk_ante_kernel<<<MSEL, 32, 0, stream>>>(fastM, top_ante, ta_fast, dout);

    // --- refinement loop (DEPTH = 2) ---
    for (int it = 0; it < 2; ++it) {
        slow_kernel<<<(MSEL * KANT * 32 + 255) / 256, 256, 0, stream>>>(
            repsF, W_a, b_a, top_ante, ta_fast, scores_a);
        attend_kernel<<<MSEL, 256, 0, stream>>>(
            repsF, scores_a, top_ante, attended, Xb);
        gemm_gate_kernel<<<(wavesBig + 7) / 8, 256, 0, stream>>>(
            Xb, WfT, b_f, attended, repsF, repsH);
    }

    // --- outputs ---
    final_kernel<<<MSEL, 64, 0, stream>>>(scores_a, top_ante, tclA, lossRows, dout);
    loss_sum_kernel<<<1, 32, 0, stream>>>(lossRows, dout);
}
